// RCIMGcn_24747601559684
// MI455X (gfx1250) — matrix operand order repaired and checked by reading, not compiled, not run on hardware
//
#include <hip/hip_runtime.h>
#include <hip/hip_fp16.h>

typedef __attribute__((ext_vector_type(16))) _Float16 v16h;
typedef __attribute__((ext_vector_type(8)))  _Float16 v8h;
typedef __attribute__((ext_vector_type(8)))  float    v8f;

#define N_USERS 8000
#define N_ENT   32000
#define EMB     64
#define N_GROUPS 4
#define N_EDGES 200000
#define INV_TAU 2.0f
#define LOG_MAX 2.0f   // logits = cos/tau bounded by 1/tau = 2 (l2-normalized inputs)

// ------------------------------------------------------------------
// Graph propagation: edge-parallel scatter-add (segment_sum semantics)
// wave covers 32 consecutive dims of one edge -> ui[e]/ei[e] are
// wave-uniform (scalar loads).
// ------------------------------------------------------------------
__global__ void __launch_bounds__(256)
scatter_kernel(const int* __restrict__ ui, const int* __restrict__ ei,
               const float* __restrict__ u_old, const float* __restrict__ e_old,
               float* __restrict__ u_new, float* __restrict__ e_new) {
  long long idx = (long long)blockIdx.x * blockDim.x + threadIdx.x;
  if (idx >= (long long)N_EDGES * EMB) return;
  int e = (int)(idx >> 6);
  int d = (int)(idx & 63);
  int u = ui[e];
  int t = ei[e];
  atomicAdd(&u_new[(long long)u * EMB + d], e_old[(long long)t * EMB + d]);
  atomicAdd(&e_new[(long long)t * EMB + d], u_old[(long long)u * EMB + d]);
}

// l2-normalize rows in place; optionally accumulate normalized rows into acc.
// One wave32 per row, 2 floats per lane.
__global__ void __launch_bounds__(256)
norm_kernel(float* __restrict__ X, float* __restrict__ acc, int nrows) {
  int row  = blockIdx.x * (blockDim.x >> 5) + (threadIdx.x >> 5);
  int lane = threadIdx.x & 31;
  if (row >= nrows) return;
  float2* xp = (float2*)(X + (long long)row * EMB);
  float2 v = xp[lane];
  float ss = v.x * v.x + v.y * v.y;
  ss += __shfl_xor(ss, 1);
  ss += __shfl_xor(ss, 2);
  ss += __shfl_xor(ss, 4);
  ss += __shfl_xor(ss, 8);
  ss += __shfl_xor(ss, 16);
  float inv = 1.0f / fmaxf(sqrtf(ss), 1e-12f);
  v.x *= inv; v.y *= inv;
  xp[lane] = v;
  if (acc) {
    float2* ap = (float2*)(acc + (long long)row * EMB);
    float2 a = ap[lane];
    a.x += v.x; a.y += v.y;
    ap[lane] = a;
  }
}

// attention softmax over groups + residual combine -> user_int_emb (f32 exact)
__global__ void __launch_bounds__(256)
attn_kernel(const float* __restrict__ acc, const float* __restrict__ user_emb,
            const float* __restrict__ user_ui, float* __restrict__ out) {
  int row  = blockIdx.x * (blockDim.x >> 5) + (threadIdx.x >> 5);
  int lane = threadIdx.x & 31;
  if (row >= N_USERS) return;
  const float2* wp = (const float2*)(user_ui + (long long)row * EMB);
  float2 w = wp[lane];
  float  s[N_GROUPS];
  float2 av[N_GROUPS];
#pragma unroll
  for (int g = 0; g < N_GROUPS; ++g) {
    const float2* ap = (const float2*)(acc + ((long long)g * N_USERS + row) * EMB);
    float2 a = ap[lane];
    av[g] = a;
    float d = a.x * w.x + a.y * w.y;
    d += __shfl_xor(d, 1);
    d += __shfl_xor(d, 2);
    d += __shfl_xor(d, 4);
    d += __shfl_xor(d, 8);
    d += __shfl_xor(d, 16);
    s[g] = d;
  }
  float m = fmaxf(fmaxf(s[0], s[1]), fmaxf(s[2], s[3]));
  float es[N_GROUPS], tot = 0.f;
#pragma unroll
  for (int g = 0; g < N_GROUPS; ++g) { es[g] = expf(s[g] - m); tot += es[g]; }
  const float2* up = (const float2*)(user_emb + (long long)row * EMB);
  float2 o = up[lane];
#pragma unroll
  for (int g = 0; g < N_GROUPS; ++g) {
    float a = es[g] / tot;
    o.x += av[g].x * a;
    o.y += av[g].y * a;
  }
  ((float2*)(out + (long long)row * EMB))[lane] = o;
}

// proj(z) = elu(z@W1+b1)@W2+b2, l2-normalized, emitted as f16 for WMMA
__global__ void __launch_bounds__(64)
proj_kernel(const float* __restrict__ Z, const float* __restrict__ W1,
            const float* __restrict__ b1, const float* __restrict__ W2,
            const float* __restrict__ b2, _Float16* __restrict__ out) {
  __shared__ float zs[EMB];
  __shared__ float hs[EMB];
  __shared__ float red[EMB];
  int r = blockIdx.x;
  int t = threadIdx.x;
  zs[t] = Z[(long long)r * EMB + t];
  __syncthreads();
  float a1 = b1[t];
#pragma unroll 8
  for (int k = 0; k < EMB; ++k) a1 = fmaf(zs[k], W1[k * EMB + t], a1);
  hs[t] = a1 > 0.f ? a1 : (expf(a1) - 1.f);   // elu, alpha=1
  __syncthreads();
  float a2 = b2[t];
#pragma unroll 8
  for (int k = 0; k < EMB; ++k) a2 = fmaf(hs[k], W2[k * EMB + t], a2);
  red[t] = a2 * a2;
  __syncthreads();
  for (int s = 32; s > 0; s >>= 1) {
    if (t < s) red[t] += red[t + s];
    __syncthreads();
  }
  float inv = 1.0f / fmaxf(sqrtf(red[0]), 1e-12f);
  out[(long long)r * EMB + t] = (_Float16)(a2 * inv);
}

// diagonal logits: diag[i] = (p1[i] . p2[i]) / tau  (wave per row)
__global__ void __launch_bounds__(256)
diag_kernel(const _Float16* __restrict__ p1h, const _Float16* __restrict__ p2h,
            float* __restrict__ diag) {
  int row  = blockIdx.x * (blockDim.x >> 5) + (threadIdx.x >> 5);
  int lane = threadIdx.x & 31;
  if (row >= N_USERS) return;
  const _Float16* a = p1h + (long long)row * EMB + lane * 2;
  const _Float16* b = p2h + (long long)row * EMB + lane * 2;
  float d = (float)a[0] * (float)b[0] + (float)a[1] * (float)b[1];
  d += __shfl_xor(d, 1);
  d += __shfl_xor(d, 2);
  d += __shfl_xor(d, 4);
  d += __shfl_xor(d, 8);
  d += __shfl_xor(d, 16);
  if (lane == 0) diag[row] = d * INV_TAU;
}

// ------------------------------------------------------------------
// Streaming InfoNCE statistics via WMMA f32<-f16 16x16x32.
// A layout (ISA 7.12.2): lanes 0-15 = rows, K {0-7,16-23}; lanes 16-31 K {8-15,24-31}
// B layout (K=32): lane -> col (lane&15), same K order as A:
//   lanes 0-15 K {0-7,16-23}; lanes 16-31 K {8-15,24-31}.
// ------------------------------------------------------------------
__device__ __forceinline__ v16h load_A(const _Float16* __restrict__ X,
                                       int row0, int lane, int kBase) {
  int row = row0 + (lane & 15);
  const _Float16* p = X + (long long)row * EMB + kBase + ((lane >> 4) << 3);
  v8h lo = *(const v8h*)p;          // K = kb .. kb+7
  v8h hi = *(const v8h*)(p + 16);   // K = kb+16 .. kb+23
  v16h a;
#pragma unroll
  for (int e = 0; e < 8; ++e) { a[e] = lo[e]; a[e + 8] = hi[e]; }
  return a;
}

__device__ __forceinline__ v16h load_B(const _Float16* __restrict__ X,
                                       int col0, int lane, int kBase) {
  int col = col0 + (lane & 15);
  const _Float16* p = X + (long long)col * EMB + kBase + ((lane >> 4) << 3);
  v8h lo = *(const v8h*)p;
  v8h hi = *(const v8h*)(p + 16);
  v16h b;
#pragma unroll
  for (int e = 0; e < 8; ++e) { b[e] = lo[e]; b[e + 8] = hi[e]; }
  return b;
}

__global__ void __launch_bounds__(128)
logits_kernel(const _Float16* __restrict__ p1h, const _Float16* __restrict__ p2h,
              float* __restrict__ rowsum, float* __restrict__ colsum) {
  const int nTiles = N_USERS / 16;          // 500, divisible by 4 waves
  int wave = threadIdx.x >> 5;
  int lane = threadIdx.x & 31;
  int r0 = blockIdx.x * 16;

  v16h a0 = load_A(p1h, r0, lane, 0);       // K 0..31
  v16h a1 = load_A(p1h, r0, lane, 32);      // K 32..63

  float rs[8];
#pragma unroll
  for (int v = 0; v < 8; ++v) rs[v] = 0.f;

  float* mycol = colsum + (lane & 15);      // both halves hit same addr: HW serializes

  for (int j = wave; j < nTiles; j += 4) {  // uniform trip count per wave
    int c0 = j * 16;
    v16h b0 = load_B(p2h, c0, lane, 0);
    v16h b1 = load_B(p2h, c0, lane, 32);
    v8f c = {};
    c = __builtin_amdgcn_wmma_f32_16x16x32_f16(false, a0, false, b0,
                                               (short)0, c, false, false);
    c = __builtin_amdgcn_wmma_f32_16x16x32_f16(false, a1, false, b1,
                                               (short)0, c, false, false);
    float csum = 0.f;
#pragma unroll
    for (int v = 0; v < 8; ++v) {
      // C layout: M = v + 8*(lane>=16), N = lane&15
      float ex = __expf(c[v] * INV_TAU - LOG_MAX);  // fixed safe shift, |logit|<=2
      rs[v] += ex;
      csum  += ex;
    }
    // all 32 lanes active (EXEC stays full; no DS round-trip, no exec juggling)
    atomicAdd(mycol + c0, csum);
  }

  // rowsum: reduce over N (16 lanes per half), combine 4 waves via LDS, store
  __shared__ float rsum[4][16];
#pragma unroll
  for (int v = 0; v < 8; ++v) {
    float s = rs[v];
    s += __shfl_xor(s, 1);
    s += __shfl_xor(s, 2);
    s += __shfl_xor(s, 4);
    s += __shfl_xor(s, 8);
    if (lane == 0)  rsum[wave][v] = s;
    if (lane == 16) rsum[wave][v + 8] = s;
  }
  __syncthreads();
  int t = threadIdx.x;
  if (t < 16)
    rowsum[r0 + t] = rsum[0][t] + rsum[1][t] + rsum[2][t] + rsum[3][t];
}

__global__ void __launch_bounds__(256)
loss_kernel(const float* __restrict__ rowsum, const float* __restrict__ colsum,
            const float* __restrict__ diag, float* __restrict__ out_loss) {
  __shared__ float red[256];
  int t = threadIdx.x;
  float acc = 0.f;
  for (int i = t; i < N_USERS; i += 256)
    acc += 0.5f * (logf(rowsum[i]) + logf(colsum[i])) + LOG_MAX - diag[i];
  red[t] = acc;
  __syncthreads();
  for (int s = 128; s > 0; s >>= 1) {
    if (t < s) red[t] += red[t + s];
    __syncthreads();
  }
  if (t == 0) *out_loss = red[0] / (float)N_USERS;
}

// ------------------------------------------------------------------
extern "C" void kernel_launch(void* const* d_in, const int* in_sizes, int n_in,
                              void* d_out, int out_size, void* d_ws, size_t ws_size,
                              hipStream_t stream) {
  (void)in_sizes; (void)n_in; (void)out_size; (void)ws_size;
  const float* user_emb    = (const float*)d_in[0];
  const float* user_ui_emb = (const float*)d_in[1];
  const float* entity_emb  = (const float*)d_in[2];
  const float* W1 = (const float*)d_in[3];
  const float* b1 = (const float*)d_in[4];
  const float* W2 = (const float*)d_in[5];
  const float* b2 = (const float*)d_in[6];
  const int* ui_index = (const int*)d_in[7];
  const int* ei_index = (const int*)d_in[8];
  // d_in[9] = mess_dropout (0 in eval mode) -> identity, unused

  float* out = (float*)d_out;  // [0..512000): user_int_emb ; [512000]: loss

  // workspace carve-out: ~30.8 MB total
  float* ws = (float*)d_ws;
  const long long U = (long long)N_USERS * EMB;   // 512000
  const long long E = (long long)N_ENT * EMB;     // 2048000
  float* u_a    = ws;  ws += U;
  float* u_b    = ws;  ws += U;
  float* e_a    = ws;  ws += E;
  float* e_b    = ws;  ws += E;
  float* acc    = ws;  ws += (long long)N_GROUPS * U;
  float* rowsum = ws;  ws += N_USERS;
  float* colsum = ws;  ws += N_USERS;
  float* diag   = ws;  ws += N_USERS;
  _Float16* p1h = (_Float16*)ws; ws += U / 2;     // 32B-aligned offset
  _Float16* p2h = (_Float16*)ws; ws += U / 2;

  (void)hipMemsetAsync(acc, 0, (size_t)N_GROUPS * U * sizeof(float), stream);
  (void)hipMemsetAsync(colsum, 0, (size_t)N_USERS * sizeof(float), stream);

  const long long scatterWork = (long long)N_EDGES * EMB;
  const int scatterBlocks = (int)((scatterWork + 255) / 256);

  for (int g = 0; g < N_GROUPS; ++g) {
    (void)hipMemcpyAsync(u_a, user_emb,  U * sizeof(float), hipMemcpyDeviceToDevice, stream);
    (void)hipMemcpyAsync(e_a, entity_emb, E * sizeof(float), hipMemcpyDeviceToDevice, stream);
    float* u_old = u_a; float* u_new = u_b;
    float* e_old = e_a; float* e_new = e_b;
    const int* ui = ui_index + (long long)g * N_EDGES;
    const int* ei = ei_index + (long long)g * N_EDGES;
    for (int hop = 0; hop < 2; ++hop) {
      (void)hipMemsetAsync(u_new, 0, U * sizeof(float), stream);
      (void)hipMemsetAsync(e_new, 0, E * sizeof(float), stream);
      scatter_kernel<<<scatterBlocks, 256, 0, stream>>>(ui, ei, u_old, e_old, u_new, e_new);
      norm_kernel<<<N_USERS / 8, 256, 0, stream>>>(u_new, acc + (long long)g * U, N_USERS);
      norm_kernel<<<N_ENT / 8, 256, 0, stream>>>(e_new, nullptr, N_ENT);
      float* t0 = u_old; u_old = u_new; u_new = t0;   // captured at record time:
      float* t1 = e_old; e_old = e_new; e_new = t1;   // deterministic
    }
  }

  attn_kernel<<<N_USERS / 8, 256, 0, stream>>>(acc, user_emb, user_ui_emb, out);

  proj_kernel<<<N_USERS, 64, 0, stream>>>(user_ui_emb, W1, b1, W2, b2, p1h);
  proj_kernel<<<N_USERS, 64, 0, stream>>>(out,         W1, b1, W2, b2, p2h);

  diag_kernel<<<N_USERS / 8, 256, 0, stream>>>(p1h, p2h, diag);

  logits_kernel<<<N_USERS / 16, 128, 0, stream>>>(p1h, p2h, rowsum, colsum);

  loss_kernel<<<1, 256, 0, stream>>>(rowsum, colsum, diag, out + U);
}
